// ModelF_Unified_25769804273
// MI455X (gfx1250) — matrix operand order repaired and checked by reading, not compiled, not run on hardware
//
#include <hip/hip_runtime.h>
#include <hip/hip_bf16.h>
#include <math.h>

// ---------------------------------------------------------------------------
// CDNA5 (gfx1250) wave32 WMMA types
// ---------------------------------------------------------------------------
typedef __attribute__((ext_vector_type(16))) __bf16 v16bf;
typedef __attribute__((ext_vector_type(8)))  float  v8f;

union Frag16 { v16bf v; unsigned u[8]; unsigned short h[16]; };

__device__ __forceinline__ unsigned short f2bf(float f) {
    union { float f; unsigned u; } c; c.f = f;
    unsigned r = c.u + 0x7FFFu + ((c.u >> 16) & 1u);   // round-to-nearest-even
    return (unsigned short)(r >> 16);
}
__device__ __forceinline__ float gelu_exact(float x) {
    return 0.5f * x * (1.0f + erff(x * 0.70710678118654752f));
}

// ---------------------------------------------------------------------------
// Generic batched WMMA GEMM:  C[bz] = act(A[bz] * B[bz] + bias[bz])
// A: [M,K] fp32 row-major (lda), B: [K,N] fp32 row-major (ldb) or [N,K] if TRANSB
// 256 threads (8 waves), block tile 128x128, K-step 32.
// Wave (wm,wn) = 4 M-subtiles x 2 N-subtiles of 16x16 -> 8 wmma per K-step.
// ---------------------------------------------------------------------------
#define BM 128
#define BN 128
#define BK 32
#define AST 34   // LDS stride (bf16 elems), pad 2 -> conflict-free
#define BST 34

template<bool TRANSB, int ACT>
__global__ __launch_bounds__(256) void gemm_wmma(
    const float* __restrict__ A, long long lda, long long Asb,
    const float* __restrict__ B, long long ldb, long long Bsb,
    const float* __restrict__ bias, long long biasSb,
    float* __restrict__ C, long long ldc, long long Csb,
    int M, int N, int K)
{
    __shared__ unsigned short As[BM * AST];
    __shared__ unsigned short Bt[BN * BST];   // Bt[n][k]

    const int tid  = threadIdx.x;
    const int lane = tid & 31, wave = tid >> 5;
    const int half = lane >> 4, l15 = lane & 15;
    const int wm = wave >> 2, wn = wave & 3;
    const int bm0 = blockIdx.y * BM;
    const int bn0 = blockIdx.x * BN;
    const int bz  = blockIdx.z;

    const float* Ab = A + (long long)bz * Asb;
    const float* Bb = B + (long long)bz * Bsb;
    float*       Cb = C + (long long)bz * Csb;

    v8f acc[4][2];
    const v8f vzero = {0.f,0.f,0.f,0.f,0.f,0.f,0.f,0.f};
    #pragma unroll
    for (int i = 0; i < 4; i++)
        #pragma unroll
        for (int j = 0; j < 2; j++) acc[i][j] = vzero;

    for (int k0 = 0; k0 < K; k0 += BK) {
        // ---- stage A (128x32) ----
        {
            int row = tid >> 1, cb = (tid & 1) * 16;
            int gm = bm0 + row;
            unsigned short* dst = &As[row * AST + cb];
            if (gm < M) {
                const float* src = Ab + (long long)gm * lda + k0 + cb;
                #pragma unroll
                for (int q = 0; q < 4; q++) {
                    float4 f = *(const float4*)(src + q * 4);
                    dst[q*4+0] = f2bf(f.x); dst[q*4+1] = f2bf(f.y);
                    dst[q*4+2] = f2bf(f.z); dst[q*4+3] = f2bf(f.w);
                }
            } else {
                #pragma unroll
                for (int q = 0; q < 16; q++) dst[q] = 0;
            }
        }
        // ---- stage B transposed into Bt[n][k] ----
        if (!TRANSB) {
            int kr = tid >> 3, nb = (tid & 7) * 16;
            const float* src = Bb + (long long)(k0 + kr) * ldb + bn0 + nb;
            #pragma unroll
            for (int q = 0; q < 4; q++) {
                float4 f = *(const float4*)(src + q * 4);
                Bt[(nb + q*4 + 0) * BST + kr] = f2bf(f.x);
                Bt[(nb + q*4 + 1) * BST + kr] = f2bf(f.y);
                Bt[(nb + q*4 + 2) * BST + kr] = f2bf(f.z);
                Bt[(nb + q*4 + 3) * BST + kr] = f2bf(f.w);
            }
        } else {
            int nr = tid >> 1, cb2 = (tid & 1) * 16;
            const float* src = Bb + (long long)(bn0 + nr) * ldb + k0 + cb2;
            unsigned short* dst = &Bt[nr * BST + cb2];
            #pragma unroll
            for (int q = 0; q < 4; q++) {
                float4 f = *(const float4*)(src + q * 4);
                dst[q*4+0] = f2bf(f.x); dst[q*4+1] = f2bf(f.y);
                dst[q*4+2] = f2bf(f.z); dst[q*4+3] = f2bf(f.w);
            }
        }
        __syncthreads();

        // ---- fragments + WMMA ----
        Frag16 a[4], b[2];
        #pragma unroll
        for (int mt = 0; mt < 4; mt++) {
            const unsigned short* ar = &As[(wm*64 + mt*16 + l15) * AST];
            #pragma unroll
            for (int j = 0; j < 4; j++) {
                a[mt].u[j]     = *(const unsigned*)(ar + half*8 + 2*j);        // K 0..7 / 8..15
                a[mt].u[4 + j] = *(const unsigned*)(ar + 16 + half*8 + 2*j);   // K 16..23 / 24..31
            }
        }
        #pragma unroll
        for (int nt = 0; nt < 2; nt++) {
            const unsigned short* br = &Bt[(wn*32 + nt*16 + l15) * BST];
            #pragma unroll
            for (int j = 0; j < 8; j++)
                b[nt].u[j] = *(const unsigned*)(br + (j>>2)*16 + half*8 + 2*(j&3));  // K 0..7 / 8..15, then 16..23 / 24..31
        }
        #pragma unroll
        for (int mt = 0; mt < 4; mt++)
            #pragma unroll
            for (int nt = 0; nt < 2; nt++)
                acc[mt][nt] = __builtin_amdgcn_wmma_f32_16x16x32_bf16(
                    false, a[mt].v, false, b[nt].v, (short)0, acc[mt][nt], false, false);
        __syncthreads();
    }

    // ---- epilogue ----
    #pragma unroll
    for (int mt = 0; mt < 4; mt++) {
        #pragma unroll
        for (int nt = 0; nt < 2; nt++) {
            int gn = bn0 + wn*32 + nt*16 + l15;
            float bv = bias ? bias[(long long)bz * biasSb + gn] : 0.0f;
            #pragma unroll
            for (int r = 0; r < 8; r++) {
                int gm = bm0 + wm*64 + mt*16 + half*8 + r;
                if (gm < M) {
                    float v = acc[mt][nt][r] + bv;
                    if (ACT == 1) v = gelu_exact(v);
                    Cb[(long long)gm * ldc + gn] = v;
                }
            }
        }
    }
}

// ---------------------------------------------------------------------------
// Fused attention per (b,h): S = QK^T/8 ; P = softmax(S) ; O = P V
// q/k/v/o layout: [B, N, H*Dh] with head offset h*64.  N=128, Dh=64.
// ---------------------------------------------------------------------------
#define QST 66    // 64 + 2 pad
#define VST 130   // 128 + 2 pad
#define PST 130

__global__ __launch_bounds__(256) void attn_kernel(
    const float* __restrict__ q, const float* __restrict__ k,
    const float* __restrict__ v, float* __restrict__ o)
{
    extern __shared__ unsigned short sm[];
    unsigned short* Qs = sm;                    // 128*66
    unsigned short* Ks = sm + 128 * QST;        // 128*66
    unsigned short* Ps = sm;                    // 128*130 (reuses Q+K region)
    unsigned short* Vt = sm + 2 * 128 * QST;    // Vt[dh][m'] 64*130

    const int tid  = threadIdx.x;
    const int lane = tid & 31, w = tid >> 5;
    const int half = lane >> 4, l15 = lane & 15;
    const int bh = blockIdx.x;
    const int b = bh >> 3, h = bh & 7;
    const long long base = (long long)b * 128 * 512 + h * 64;

    // ---- stage Q, K (row-major) and V transposed ----
    {
        int m = tid >> 1, cb = (tid & 1) * 32;
        const float* qs = q + base + (long long)m * 512 + cb;
        const float* ks = k + base + (long long)m * 512 + cb;
        const float* vs = v + base + (long long)m * 512 + cb;
        #pragma unroll
        for (int j = 0; j < 8; j++) {
            float4 fq = *(const float4*)(qs + j*4);
            float4 fk = *(const float4*)(ks + j*4);
            float4 fv = *(const float4*)(vs + j*4);
            int c0 = cb + j*4;
            Qs[m*QST + c0+0] = f2bf(fq.x); Qs[m*QST + c0+1] = f2bf(fq.y);
            Qs[m*QST + c0+2] = f2bf(fq.z); Qs[m*QST + c0+3] = f2bf(fq.w);
            Ks[m*QST + c0+0] = f2bf(fk.x); Ks[m*QST + c0+1] = f2bf(fk.y);
            Ks[m*QST + c0+2] = f2bf(fk.z); Ks[m*QST + c0+3] = f2bf(fk.w);
            Vt[(c0+0)*VST + m] = f2bf(fv.x); Vt[(c0+1)*VST + m] = f2bf(fv.y);
            Vt[(c0+2)*VST + m] = f2bf(fv.z); Vt[(c0+3)*VST + m] = f2bf(fv.w);
        }
    }
    __syncthreads();

    // ---- S = Q K^T : wave w owns rows [16w,16w+16), all 8 column tiles ----
    v8f s[8];
    const v8f vzero = {0.f,0.f,0.f,0.f,0.f,0.f,0.f,0.f};
    #pragma unroll
    for (int nt = 0; nt < 8; nt++) s[nt] = vzero;

    #pragma unroll
    for (int k0 = 0; k0 < 64; k0 += 32) {
        Frag16 a;
        const unsigned short* ar = &Qs[(w*16 + l15) * QST + k0];
        #pragma unroll
        for (int j = 0; j < 4; j++) {
            a.u[j]     = *(const unsigned*)(ar + half*8 + 2*j);
            a.u[4 + j] = *(const unsigned*)(ar + 16 + half*8 + 2*j);
        }
        #pragma unroll
        for (int nt = 0; nt < 8; nt++) {
            Frag16 bf;
            const unsigned short* br = &Ks[(nt*16 + l15) * QST + k0];  // Ks == B^T[n][k]
            #pragma unroll
            for (int j = 0; j < 8; j++)
                bf.u[j] = *(const unsigned*)(br + (j>>2)*16 + half*8 + 2*(j&3));
            s[nt] = __builtin_amdgcn_wmma_f32_16x16x32_bf16(
                false, a.v, false, bf.v, (short)0, s[nt], false, false);
        }
    }
    __syncthreads();   // all Qs/Ks reads done before Ps overwrite

    // ---- softmax in registers; row m = w*16 + half*8 + r ----
    const float scale = 0.125f;  // 1/sqrt(64)
    #pragma unroll
    for (int r = 0; r < 8; r++) {
        float rm = -3.4e38f;
        #pragma unroll
        for (int nt = 0; nt < 8; nt++) rm = fmaxf(rm, s[nt][r]);
        rm = fmaxf(rm, __shfl_xor(rm, 1, 32));
        rm = fmaxf(rm, __shfl_xor(rm, 2, 32));
        rm = fmaxf(rm, __shfl_xor(rm, 4, 32));
        rm = fmaxf(rm, __shfl_xor(rm, 8, 32));
        float sum = 0.0f;
        #pragma unroll
        for (int nt = 0; nt < 8; nt++) {
            float e = expf(scale * (s[nt][r] - rm));
            s[nt][r] = e;
            sum += e;
        }
        sum += __shfl_xor(sum, 1, 32);
        sum += __shfl_xor(sum, 2, 32);
        sum += __shfl_xor(sum, 4, 32);
        sum += __shfl_xor(sum, 8, 32);
        float inv = 1.0f / sum;
        int row = w*16 + half*8 + r;
        #pragma unroll
        for (int nt = 0; nt < 8; nt++)
            Ps[row * PST + nt*16 + l15] = f2bf(s[nt][r] * inv);
    }

    // ---- O = P V : wave w rows [16w,16w+16), 4 Dh tiles, K=128 ----
    v8f oa[4];
    #pragma unroll
    for (int nt = 0; nt < 4; nt++) oa[nt] = vzero;

    #pragma unroll
    for (int k0 = 0; k0 < 128; k0 += 32) {
        Frag16 a;
        const unsigned short* ar = &Ps[(w*16 + l15) * PST + k0];
        #pragma unroll
        for (int j = 0; j < 4; j++) {
            a.u[j]     = *(const unsigned*)(ar + half*8 + 2*j);
            a.u[4 + j] = *(const unsigned*)(ar + 16 + half*8 + 2*j);
        }
        #pragma unroll
        for (int nt = 0; nt < 4; nt++) {
            Frag16 bf;
            const unsigned short* br = &Vt[(nt*16 + l15) * VST + k0];  // Vt == B^T[dh][m']
            #pragma unroll
            for (int j = 0; j < 8; j++)
                bf.u[j] = *(const unsigned*)(br + (j>>2)*16 + half*8 + 2*(j&3));
            oa[nt] = __builtin_amdgcn_wmma_f32_16x16x32_bf16(
                false, a.v, false, bf.v, (short)0, oa[nt], false, false);
        }
    }

    #pragma unroll
    for (int nt = 0; nt < 4; nt++)
        #pragma unroll
        for (int r = 0; r < 8; r++) {
            int row = w*16 + half*8 + r;
            o[base + (long long)row * 512 + nt*16 + l15] = oa[nt][r];
        }
}

// ---------------------------------------------------------------------------
// encoder: x[row][d] = sum_j feats_j * W[j][d] + b[d]
// ---------------------------------------------------------------------------
__global__ void encoder_kernel(const float* __restrict__ bs, const float* __restrict__ ins,
                               const float* __restrict__ tr, const float* __restrict__ im,
                               const float* __restrict__ W, const float* __restrict__ bvec,
                               float* __restrict__ x)
{
    int row = blockIdx.x;
    float f0 = bs[row], f1 = ins[row], f2 = tr[row], f3 = im[row];
    float* xr = x + (long long)row * 512;
    for (int d = threadIdx.x; d < 512; d += blockDim.x)
        xr[d] = f0 * W[d] + f1 * W[512 + d] + f2 * W[1024 + d] + f3 * W[1536 + d] + bvec[d];
}

// ---------------------------------------------------------------------------
// y = LayerNorm(x + t) * g + b   over rows of width 512
// ---------------------------------------------------------------------------
__global__ __launch_bounds__(256) void add_ln_kernel(
    const float* __restrict__ x, const float* __restrict__ t,
    const float* __restrict__ g, const float* __restrict__ bb,
    float* __restrict__ y)
{
    __shared__ float red[256];
    int row = blockIdx.x, tid = threadIdx.x;
    const float* xr = x + (long long)row * 512;
    const float* tr = t + (long long)row * 512;
    float v0 = xr[tid] + tr[tid];
    float v1 = xr[tid + 256] + tr[tid + 256];
    red[tid] = v0 + v1; __syncthreads();
    for (int off = 128; off > 0; off >>= 1) { if (tid < off) red[tid] += red[tid + off]; __syncthreads(); }
    float mean = red[0] * (1.0f / 512.0f);
    __syncthreads();
    float d0 = v0 - mean, d1 = v1 - mean;
    red[tid] = d0 * d0 + d1 * d1; __syncthreads();
    for (int off = 128; off > 0; off >>= 1) { if (tid < off) red[tid] += red[tid + off]; __syncthreads(); }
    float inv = rsqrtf(red[0] * (1.0f / 512.0f) + 1e-5f);
    float* yr = y + (long long)row * 512;
    yr[tid]       = d0 * inv * g[tid] + bb[tid];
    yr[tid + 256] = d1 * inv * g[tid + 256] + bb[tid + 256];
}

// ---------------------------------------------------------------------------
// hyper-net: mod[b] = gelu(emb[idx[b]] @ W1 + b1) @ W2 + b2
// ---------------------------------------------------------------------------
__global__ __launch_bounds__(64) void hyper_kernel(
    const int* __restrict__ idx, const float* __restrict__ emb,
    const float* __restrict__ W1, const float* __restrict__ b1,
    const float* __restrict__ W2, const float* __restrict__ b2,
    float* __restrict__ mod)
{
    __shared__ float esh[32];
    __shared__ float hsh[64];
    int b = blockIdx.x, tid = threadIdx.x;
    int id = idx[b];
    if (tid < 32) esh[tid] = emb[(long long)id * 32 + tid];
    __syncthreads();
    float acc = b1[tid];
    for (int i = 0; i < 32; i++) acc += esh[i] * W1[i * 64 + tid];
    hsh[tid] = gelu_exact(acc);
    __syncthreads();
    for (int d = tid; d < 512; d += 64) {
        float a2 = b2[d];
        for (int j = 0; j < 64; j++) a2 += hsh[j] * W2[j * 512 + d];
        mod[(long long)b * 512 + d] = a2;
    }
}

// xm[b,n,d] = x[b,n,d] * mod[b,d]
__global__ void mod_mul_kernel(const float* __restrict__ x, const float* __restrict__ mod,
                               float* __restrict__ xm)
{
    long long i = (long long)blockIdx.x * blockDim.x + threadIdx.x;  // < 64*128*512
    int b = (int)(i >> 16);       // 128*512 = 65536
    int d = (int)(i & 511);
    xm[i] = x[i] * mod[b * 512 + d];
}

// deltas[b,n] = h[n][b][:] . exp_W2[n] + exp_b2[n]   (one wave per dot)
__global__ __launch_bounds__(256) void deltas_kernel(
    const float* __restrict__ hbuf, const float* __restrict__ W2,
    const float* __restrict__ b2, float* __restrict__ out)
{
    int tid = threadIdx.x, lane = tid & 31, w = tid >> 5;
    long long rid = (long long)blockIdx.x * 8 + w;  // b*128+n
    int b = (int)(rid >> 7), n = (int)(rid & 127);
    const float* hr = hbuf + ((long long)n * 64 + b) * 512;
    const float* wr = W2 + (long long)n * 512;
    float acc = 0.0f;
    for (int d = lane; d < 512; d += 32) acc += hr[d] * wr[d];
    acc += __shfl_xor(acc, 16, 32);
    acc += __shfl_xor(acc, 8, 32);
    acc += __shfl_xor(acc, 4, 32);
    acc += __shfl_xor(acc, 2, 32);
    acc += __shfl_xor(acc, 1, 32);
    if (lane == 0) out[rid] = acc + b2[n];
}

// adj = sigmoid(logits + g_noise)   (TAU = 1)
__global__ void adj_kernel(const float* __restrict__ logits, const float* __restrict__ gn,
                           float* __restrict__ adj)
{
    long long i = (long long)blockIdx.x * blockDim.x + threadIdx.x;
    float z = logits[i] + gn[i];
    adj[i] = 1.0f / (1.0f + expf(-z));
}

// ---------------------------------------------------------------------------
extern "C" void kernel_launch(void* const* d_in, const int* in_sizes, int n_in,
                              void* d_out, int out_size, void* d_ws, size_t ws_size,
                              hipStream_t stream)
{
    (void)in_sizes; (void)n_in; (void)out_size; (void)ws_size;

    const float* base_s = (const float*)d_in[0];
    const float* int_s  = (const float*)d_in[1];
    const float* tgt    = (const float*)d_in[2];
    const float* mask   = (const float*)d_in[3];
    const int*   nidx   = (const int*)  d_in[4];
    const float* gnz    = (const float*)d_in[5];
    const float* enc_W  = (const float*)d_in[6];
    const float* enc_b  = (const float*)d_in[7];
    const float* Wq = (const float*)d_in[8],  *bq = (const float*)d_in[9];
    const float* Wk = (const float*)d_in[10], *bk = (const float*)d_in[11];
    const float* Wv = (const float*)d_in[12], *bv = (const float*)d_in[13];
    const float* Wo = (const float*)d_in[14], *bo = (const float*)d_in[15];
    const float* ln1g = (const float*)d_in[16], *ln1b = (const float*)d_in[17];
    const float* Wf1 = (const float*)d_in[18], *bf1 = (const float*)d_in[19];
    const float* Wf2 = (const float*)d_in[20], *bf2 = (const float*)d_in[21];
    const float* ln2g = (const float*)d_in[22], *ln2b = (const float*)d_in[23];
    const float* emb = (const float*)d_in[24];
    const float* hW1 = (const float*)d_in[25], *hb1 = (const float*)d_in[26];
    const float* hW2 = (const float*)d_in[27], *hb2 = (const float*)d_in[28];
    const float* eW1 = (const float*)d_in[29], *eb1 = (const float*)d_in[30];
    const float* eW2 = (const float*)d_in[31], *eb2 = (const float*)d_in[32];
    const float* dpW = (const float*)d_in[33], *dpb = (const float*)d_in[34];
    const float* dcW = (const float*)d_in[35], *dcb = (const float*)d_in[36];

    const long long TOK = 8192;           // B*N
    const long long TD  = TOK * 512;      // 4,194,304
    const long long DD  = 512 * 512;

    float* xw = (float*)d_ws;
    float* qb = xw + TD;
    float* kb = qb + TD;
    float* vb = kb + TD;
    float* ob = vb + TD;
    float* tb = ob + TD;
    float* hb = tb + TD;                  // 8192*2048
    float* modb = hb + TOK * 2048;        // 64*512
    // buffer reuse after the transformer stack:
    float* pb  = qb;                      // DAG parent
    float* cb  = kb;                      // DAG child
    float* xmb = vb;                      // modulated x
    float* ehb = ob;                      // expert hidden [node][b][d]

    const size_t ATT_SMEM = (size_t)(2 * 128 * QST + 64 * VST) * sizeof(unsigned short); // 50432

    // 1) encoder
    encoder_kernel<<<8192, 128, 0, stream>>>(base_s, int_s, tgt, mask, enc_W, enc_b, xw);

    // 2) transformer stack
    dim3 g512(4, 64, 1), g2048(16, 64, 1);
    for (int l = 0; l < 4; l++) {
        const float* wq = Wq + (long long)l * DD; const float* bql = bq + l * 512;
        const float* wk = Wk + (long long)l * DD; const float* bkl = bk + l * 512;
        const float* wv = Wv + (long long)l * DD; const float* bvl = bv + l * 512;
        const float* wo = Wo + (long long)l * DD; const float* bol = bo + l * 512;

        gemm_wmma<false,0><<<g512, 256, 0, stream>>>(xw,512,0, wq,512,0, bql,0, qb,512,0, 8192,512,512);
        gemm_wmma<false,0><<<g512, 256, 0, stream>>>(xw,512,0, wk,512,0, bkl,0, kb,512,0, 8192,512,512);
        gemm_wmma<false,0><<<g512, 256, 0, stream>>>(xw,512,0, wv,512,0, bvl,0, vb,512,0, 8192,512,512);

        attn_kernel<<<512, 256, ATT_SMEM, stream>>>(qb, kb, vb, ob);

        gemm_wmma<false,0><<<g512, 256, 0, stream>>>(ob,512,0, wo,512,0, bol,0, tb,512,0, 8192,512,512);
        add_ln_kernel<<<8192, 256, 0, stream>>>(xw, tb, ln1g + l*512, ln1b + l*512, xw);

        gemm_wmma<false,1><<<g2048, 256, 0, stream>>>(xw,512,0, Wf1 + (long long)l*512*2048,2048,0,
                                                      bf1 + l*2048,0, hb,2048,0, 8192,2048,512);
        gemm_wmma<false,0><<<g512, 256, 0, stream>>>(hb,2048,0, Wf2 + (long long)l*2048*512,512,0,
                                                     bf2 + l*512,0, tb,512,0, 8192,512,2048);
        add_ln_kernel<<<8192, 256, 0, stream>>>(xw, tb, ln2g + l*512, ln2b + l*512, xw);
    }

    // 3) hyper-net modulation
    hyper_kernel<<<64, 64, 0, stream>>>(nidx, emb, hW1, hb1, hW2, hb2, modb);
    mod_mul_kernel<<<16384, 256, 0, stream>>>(xw, modb, xmb);

    // 4) per-node experts: h[n] = gelu(xm[:,n,:] @ eW1[n] + eb1[n]),  M=64 per node
    gemm_wmma<false,1><<<dim3(4, 1, 128), 256, 0, stream>>>(
        xmb, 65536, 512,          // A: row stride N*D, node stride D
        eW1, 512, 262144,         // B: [node][512][512]
        eb1, 512,                 // bias per node
        ehb, 512, 32768,          // C: [node][64][512]
        64, 512, 512);

    float* out_deltas = (float*)d_out;
    float* out_logits = out_deltas + 8192;
    float* out_adj    = out_logits + (long long)64 * 128 * 128;

    deltas_kernel<<<1024, 256, 0, stream>>>(ehb, eW2, eb2, out_deltas);

    // 5) DAG heads + logits + gumbel-sigmoid
    gemm_wmma<false,0><<<g512, 256, 0, stream>>>(xw,512,0, dpW,512,0, dpb,0, pb,512,0, 8192,512,512);
    gemm_wmma<false,0><<<g512, 256, 0, stream>>>(xw,512,0, dcW,512,0, dcb,0, cb,512,0, 8192,512,512);

    gemm_wmma<true,0><<<dim3(1, 1, 64), 256, 0, stream>>>(
        pb, 512, 65536,           // A: p[b] is [128,512]
        cb, 512, 65536,           // B^T: c[b] is [128,512], used transposed
        nullptr, 0,
        out_logits, 128, 16384,
        128, 128, 512);

    adj_kernel<<<4096, 256, 0, stream>>>(out_logits, gnz, out_adj);
}
